// expirimental_RNN_77266461655423
// MI455X (gfx1250) — compile-verified, matrix operand order checked by reading, not run on hardware
//
#include <hip/hip_runtime.h>
#include <math.h>

#define B_ 64
#define T_ 512
#define I_ 128
#define H_ 1024
#define O_ 64
#define DT_ 0.1f
#define KC_ (H_ + I_)   // 1152: concatenated K (recurrent + input)

typedef __attribute__((ext_vector_type(16))) __bf16 v16bf;
typedef __attribute__((ext_vector_type(8)))  __bf16 v8bf;
typedef __attribute__((ext_vector_type(8)))  float  v8f;

static __device__ __forceinline__ v16bf mk16(v8bf lo, v8bf hi) {
  v16bf r;
#pragma unroll
  for (int i = 0; i < 8; ++i) { r[i] = lo[i]; r[i + 8] = hi[i]; }
  return r;
}

// Load one 16x32 bf16 A-fragment (or mirror 32x16 B-fragment) for
// v_wmma_f32_16x16x32_bf16 from a row-major bf16 matrix with row stride `ld`.
// Lane l (l<16) holds row (row0 + l), K = {k0..k0+7, k0+16..k0+23};
// lanes 16..31 hold the same rows, K shifted by +8. Two 16B loads per lane.
static __device__ __forceinline__ v16bf load_frag(const __bf16* __restrict__ base,
                                                  int row, int ld, int k0, int lane) {
  const __bf16* p = base + (size_t)row * ld + k0 + ((lane >> 4) << 3);
  v8bf lo = *(const v8bf*)(p);
  v8bf hi = *(const v8bf*)(p + 16);
  return mk16(lo, hi);
}

// ---------------- fused per-step kernel ----------------
// Blocks 0..31 (256 waves): recurrent GEMM + Euler update for step t:
//   h_next[m,n] = h + DT*inv_tau[n]*( (r@Whh^T + x@Wih^T)[m,n] + b_hh[n] - h )
//   r_next = tanh(h_next) stored bf16 (A operand of step t+1 / out t)
// Blocks 32..33 (16 waves): output projection of the PREVIOUS step
//   outs[:, t_out, :] = r_cur @ Who^T + b_ho   (r_cur == tanh(h) of step t_out+1)
__global__ __launch_bounds__(256) void step_kernel(
    const __bf16* __restrict__ r_cur,   // [B_,H_] bf16, = tanh(h_cur)
    const __bf16* __restrict__ x_t,     // [B_,I_] bf16, this timestep
    const float*  __restrict__ h_cur,   // [B_,H_]
    const __bf16* __restrict__ Wc,      // [H_,KC_] = [W_hh | W_ih] bf16
    const float*  __restrict__ b_hh,    // [H_]
    const float*  __restrict__ inv_tau, // [H_]
    float*        __restrict__ h_next,  // [B_,H_]
    __bf16*       __restrict__ r_next,  // [B_,H_] = tanh(h_next)
    const __bf16* __restrict__ Who,     // [O_,H_] bf16
    const float*  __restrict__ b_ho,    // [O_]
    float*        __restrict__ outs,    // [B_,T_,O_]
    int t_out)                          // previous step index, -1 on first step
{
  const int lane = threadIdx.x & 31;
  const int wid  = threadIdx.x >> 5;

  if (blockIdx.x < 32) {
    // ---- recurrent GEMM: 256 tiles (4 Mtiles x 64 Ntiles), K = 1152 ----
    const int w  = blockIdx.x * 8 + wid;   // 0..255
    const int mt = (w & 3) * 16;           // M tile origin (batch rows)
    const int nt = (w >> 2) * 16;          // N tile origin (hidden cols)

    const int arow = mt + (lane & 15);
    const int brow = nt + (lane & 15);

    v8f c = {};
#pragma unroll 4
    for (int k0 = 0; k0 < H_; k0 += 32) {  // recurrent part, K=1024
      v16bf a = load_frag(r_cur, arow, H_,  k0, lane);
      v16bf b = load_frag(Wc,    brow, KC_, k0, lane);
      c = __builtin_amdgcn_wmma_f32_16x16x32_bf16(false, a, false, b,
                                                  (short)0, c, false, false);
    }
#pragma unroll
    for (int k0 = 0; k0 < I_; k0 += 32) {  // input part, K=128
      v16bf a = load_frag(x_t, arow, I_,  k0, lane);
      v16bf b = load_frag(Wc,  brow, KC_, H_ + k0, lane);
      c = __builtin_amdgcn_wmma_f32_16x16x32_bf16(false, a, false, b,
                                                  (short)0, c, false, false);
    }

    const int   n    = nt + (lane & 15);
    const float bias = b_hh[n];
    const float itau = inv_tau[n];
#pragma unroll
    for (int v = 0; v < 8; ++v) {
      const int   m    = mt + v + ((lane >> 4) << 3);
      const float hold = h_cur[m * H_ + n];
      const float hnew = hold + DT_ * itau * (c[v] + bias - hold);
      h_next[m * H_ + n] = hnew;
      r_next[m * H_ + n] = (__bf16)tanhf(hnew);
    }
  } else {
    // ---- output projection of previous step: 16 tiles, K = 1024 ----
    if (t_out < 0) return;                       // wave-uniform: EXEC stays full
    const int w2 = (blockIdx.x - 32) * 8 + wid;  // 0..15
    const int mt = (w2 & 3) * 16;
    const int ot = (w2 >> 2) * 16;

    const int arow = mt + (lane & 15);
    const int brow = ot + (lane & 15);

    v8f c = {};
#pragma unroll 4
    for (int k0 = 0; k0 < H_; k0 += 32) {
      v16bf a = load_frag(r_cur, arow, H_, k0, lane);
      v16bf b = load_frag(Who,   brow, H_, k0, lane);
      c = __builtin_amdgcn_wmma_f32_16x16x32_bf16(false, a, false, b,
                                                  (short)0, c, false, false);
    }

    const int   o    = ot + (lane & 15);
    const float bias = b_ho[o];
#pragma unroll
    for (int v = 0; v < 8; ++v) {
      const int m = mt + v + ((lane >> 4) << 3);
      outs[(size_t)m * (T_ * O_) + (size_t)t_out * O_ + o] = c[v] + bias;
    }
  }
}

// ---------------- standalone output projection (final step) ----------------
__global__ __launch_bounds__(512) void out_kernel(
    const __bf16* __restrict__ r_new,  // [B_,H_] bf16
    const __bf16* __restrict__ Who,    // [O_,H_] bf16
    const float*  __restrict__ b_ho,   // [O_]
    float*        __restrict__ outs,   // [B_,T_,O_]
    int t)
{
  const int lane = threadIdx.x & 31;
  const int wid  = threadIdx.x >> 5;   // 0..15
  const int mt   = (wid & 3) * 16;
  const int ot   = (wid >> 2) * 16;

  const int arow = mt + (lane & 15);
  const int brow = ot + (lane & 15);

  v8f c = {};
#pragma unroll 4
  for (int k0 = 0; k0 < H_; k0 += 32) {
    v16bf a = load_frag(r_new, arow, H_, k0, lane);
    v16bf b = load_frag(Who,   brow, H_, k0, lane);
    c = __builtin_amdgcn_wmma_f32_16x16x32_bf16(false, a, false, b,
                                                (short)0, c, false, false);
  }

  const int   o    = ot + (lane & 15);
  const float bias = b_ho[o];
#pragma unroll
  for (int v = 0; v < 8; ++v) {
    const int m = mt + v + ((lane >> 4) << 3);
    outs[(size_t)m * (T_ * O_) + (size_t)t * O_ + o] = c[v] + bias;
  }
}

// ---------------- one-time setup kernels ----------------
__global__ void pack_wc(const float* __restrict__ Whh, const float* __restrict__ Wih,
                        __bf16* __restrict__ Wc) {
  int idx = blockIdx.x * blockDim.x + threadIdx.x;
  if (idx >= H_ * KC_) return;
  int j = idx / KC_, k = idx - j * KC_;
  float v = (k < H_) ? Whh[j * H_ + k] : Wih[j * I_ + (k - H_)];
  Wc[idx] = (__bf16)v;
}

__global__ void pack_who(const float* __restrict__ Wf, __bf16* __restrict__ Wb) {
  int idx = blockIdx.x * blockDim.x + threadIdx.x;
  if (idx < O_ * H_) Wb[idx] = (__bf16)Wf[idx];
}

// inputs [B,T,I] f32 -> Xbf [T,B,I] bf16 (per-step slices contiguous)
__global__ void pack_x(const float* __restrict__ inputs, __bf16* __restrict__ Xbf) {
  int idx = blockIdx.x * blockDim.x + threadIdx.x;
  if (idx >= B_ * T_ * I_) return;
  int b = idx / (T_ * I_);
  int rem = idx - b * (T_ * I_);
  int t = rem / I_;
  int i = rem - t * I_;
  Xbf[(size_t)t * (B_ * I_) + b * I_ + i] = (__bf16)inputs[idx];
}

__global__ void init_h(const float* __restrict__ hidden, const float* __restrict__ taus,
                       float* __restrict__ h0, __bf16* __restrict__ r0,
                       float* __restrict__ inv_tau) {
  int idx = blockIdx.x * blockDim.x + threadIdx.x;
  if (idx < B_ * H_) {
    float h = hidden[idx];
    h0[idx] = h;
    r0[idx] = (__bf16)tanhf(h);
  }
  if (idx < H_) inv_tau[idx] = 1.0f / taus[idx];
}

__global__ void copy_h(const float* __restrict__ src, float* __restrict__ dst) {
  int idx = blockIdx.x * blockDim.x + threadIdx.x;
  if (idx < B_ * H_) dst[idx] = src[idx];
}

// ---------------- host ----------------
extern "C" void kernel_launch(void* const* d_in, const int* in_sizes, int n_in,
                              void* d_out, int out_size, void* d_ws, size_t ws_size,
                              hipStream_t stream) {
  (void)in_sizes; (void)n_in; (void)out_size; (void)ws_size;
  const float* inputs = (const float*)d_in[0];  // [B,T,I]
  const float* hidden = (const float*)d_in[1];  // [B,H]
  const float* W_ih   = (const float*)d_in[2];  // [H,I]
  const float* W_hh   = (const float*)d_in[3];  // [H,H]
  const float* b_hh   = (const float*)d_in[4];  // [H]
  const float* W_ho   = (const float*)d_in[5];  // [O,H]
  const float* b_ho   = (const float*)d_in[6];  // [O]
  const float* taus   = (const float*)d_in[7];  // [H]

  float* out         = (float*)d_out;
  float* h_final_out = out;            // B*H floats
  float* outs        = out + B_ * H_;  // B*T*O floats

  char* ws = (char*)d_ws;
  auto alloc = [&](size_t bytes) -> char* {
    char* p = ws;
    ws += (bytes + 255) & ~(size_t)255;
    return p;
  };
  __bf16* Wc    = (__bf16*)alloc((size_t)H_ * KC_ * 2);      // 2.25 MB
  __bf16* Who   = (__bf16*)alloc((size_t)O_ * H_ * 2);       // 128 KB
  __bf16* Xbf   = (__bf16*)alloc((size_t)T_ * B_ * I_ * 2);  // 8 MB
  float*  hbuf0 = (float*)alloc((size_t)B_ * H_ * 4);
  float*  hbuf1 = (float*)alloc((size_t)B_ * H_ * 4);
  __bf16* rbuf0 = (__bf16*)alloc((size_t)B_ * H_ * 2);
  __bf16* rbuf1 = (__bf16*)alloc((size_t)B_ * H_ * 2);
  float*  itau  = (float*)alloc((size_t)H_ * 4);

  pack_wc <<<(H_ * KC_ + 255) / 256, 256, 0, stream>>>(W_hh, W_ih, Wc);
  pack_who<<<(O_ * H_ + 255) / 256, 256, 0, stream>>>(W_ho, Who);
  pack_x  <<<(B_ * T_ * I_ + 255) / 256, 256, 0, stream>>>(inputs, Xbf);
  init_h  <<<(B_ * H_ + 255) / 256, 256, 0, stream>>>(hidden, taus, hbuf0, rbuf0, itau);

  float*  hc = hbuf0; float*  hn = hbuf1;
  __bf16* rc = rbuf0; __bf16* rn = rbuf1;
  for (int t = 0; t < T_; ++t) {
    // blocks 0..31: step t GEMM+update; blocks 32..33: output of step t-1
    step_kernel<<<34, 256, 0, stream>>>(rc, Xbf + (size_t)t * B_ * I_, hc,
                                        Wc, b_hh, itau, hn, rn,
                                        Who, b_ho, outs, t - 1);
    { float*  tp = hc; hc = hn; hn = tp; }
    { __bf16* tp = rc; rc = rn; rn = tp; }
  }
  out_kernel<<<1, 512, 0, stream>>>(rc, Who, b_ho, outs, T_ - 1);
  copy_h<<<(B_ * H_ + 255) / 256, 256, 0, stream>>>(hc, h_final_out);
}
